// ImmuneMemoryModule_30245159698913
// MI455X (gfx1250) — matrix operand order repaired and checked by reading, not compiled, not run on hardware
//
#include <hip/hip_runtime.h>
#include <hip/hip_bf16.h>
#include <math.h>

typedef __attribute__((ext_vector_type(16))) __bf16 bf16x16;
typedef __attribute__((ext_vector_type(8)))  __bf16 bf16x8;
typedef __attribute__((ext_vector_type(8)))  float  f32x8;

#define BATCH  65536
#define D_DIM  512
#define F_DIM  10
#define H_DIM  512
#define M_DIM  256
#define N_MEM  1024
#define THRESH 0.7f
#define SIM_EPS 1e-8f

// workspace layout (bytes) -- bmatT and w1T adjacent so they stage as one 48KB panel
#define WS_BMATT 0                                   // [16][512] bf16 (cols 10..15 zero)
#define WS_W1T   (WS_BMATT + 16*512*2)               // [512][32] bf16 (K 20..31 zero)
#define WS_W2T   (WS_W1T   + 512*32*2)               // [256][512] bf16
#define WS_BANKT (WS_W2T   + 256*512*2)              // [1024][256] bf16
#define WS_MN    (WS_BANKT + 1024*256*2)             // [1024] float

// ---------------- prep: transpose+convert weights to bf16 N-major ----------------
__global__ void prep_weights(const float* __restrict__ Bm, const float* __restrict__ W1,
                             const float* __restrict__ W2, const float* __restrict__ bank,
                             __bf16* __restrict__ bmatT, __bf16* __restrict__ w1T,
                             __bf16* __restrict__ w2T,  __bf16* __restrict__ bankT) {
    const int total = 16*512 + 512*32 + 256*512 + 1024*256;
    for (int i = blockIdx.x * blockDim.x + threadIdx.x; i < total; i += gridDim.x * blockDim.x) {
        int t = i;
        if (t < 16*512) {                       // B_mat [512,10] -> bmatT [16][512]
            int n = t / 512, k = t % 512;
            bmatT[t] = (__bf16)((n < F_DIM) ? Bm[k * F_DIM + n] : 0.0f);
            continue;
        }
        t -= 16*512;
        if (t < 512*32) {                       // W1 [20,512] -> w1T [512][32]
            int n = t / 32, k = t % 32;
            w1T[t] = (__bf16)((k < 2*F_DIM) ? W1[k * H_DIM + n] : 0.0f);
            continue;
        }
        t -= 512*32;
        if (t < 256*512) {                      // W2 [512,256] -> w2T [256][512]
            int n = t / 512, k = t % 512;
            w2T[t] = (__bf16)W2[k * M_DIM + n];
            continue;
        }
        t -= 256*512;
        bankT[t] = (__bf16)bank[t];             // memory_bank already N-major [1024][256]
    }
}

// one wave per memory row: ||memory_bank[row]||
__global__ void prep_mn(const float* __restrict__ bank, float* __restrict__ mn) {
    int lane = threadIdx.x & 31;
    int row  = blockIdx.x * (blockDim.x >> 5) + (threadIdx.x >> 5);
    if (row >= N_MEM) return;
    const float* p = bank + (size_t)row * M_DIM;
    float s = 0.0f;
    for (int j = lane; j < M_DIM; j += 32) { float v = p[j]; s += v * v; }
    #pragma unroll
    for (int xm = 16; xm >= 1; xm >>= 1) s += __shfl_xor(s, xm, 32);
    if (lane == 0) mn[row] = sqrtf(s);
}

// ---------------- fused pipeline ----------------
// LDS: 8 waves x 25KB private staging (H 16x512 + Q 16x256 + FF 16x32, bf16)
//      + two 32KB block-shared B-panel buffers (double-buffered async staging)
#define WAVE_SMEM  25600
#define CHUNK_OFF  (8 * WAVE_SMEM)        // 204800
#define SUB_SZ     32768                  // one pipeline buffer
#define TOTAL_SMEM (CHUNK_OFF + 2 * SUB_SZ) // 270336 < 320KB WGP LDS

// CDNA5 async global->LDS copy (ASYNCcnt path); each lane moves 16B per issue.
__device__ __forceinline__ void issue_async(char* lds_dst, const char* gsrc,
                                            int tid, int nbytes) {
    for (int off = tid * 16; off < nbytes; off += 256 * 16) {
        uint32_t l = (uint32_t)(uintptr_t)(lds_dst + off);   // flat LDS addr[31:0] == LDS offset
        const void* g = (const void*)(gsrc + off);
        asm volatile("global_load_async_to_lds_b128 %0, %1, off"
                     :: "v"(l), "v"(g) : "memory");
    }
}
__device__ __forceinline__ void wait_async_zero() {
    asm volatile("s_wait_asynccnt 0x0" ::: "memory");
}

__device__ __forceinline__ bf16x16 load_b_frag(const __bf16* __restrict__ base, int ldk,
                                               int n, int kbase, int half) {
    // B layout 32x16 bf16: col n; half 0: K 0-7 & 16-23, half 1: K 8-15 & 24-31
    const __bf16* p = base + (size_t)n * ldk + kbase + half * 8;
    bf16x8 lo = *(const bf16x8*)p;
    bf16x8 hi = *(const bf16x8*)(p + 16);
    return __builtin_shufflevector(lo, hi, 0,1,2,3,4,5,6,7,8,9,10,11,12,13,14,15);
}

__device__ __forceinline__ bf16x16 load_a_lds(const __bf16* __restrict__ base, int ldk,
                                              int m, int kbase, int half) {
    // A layout 16x32 bf16: lane row m; half 0: K 0-7 & 16-23, half 1: K 8-15 & 24-31
    const __bf16* p = base + (size_t)m * ldk + kbase + half * 8;
    bf16x8 lo = *(const bf16x8*)p;
    bf16x8 hi = *(const bf16x8*)(p + 16);
    return __builtin_shufflevector(lo, hi, 0,1,2,3,4,5,6,7,8,9,10,11,12,13,14,15);
}

__global__ __launch_bounds__(256)
void immune_fused(const float* __restrict__ query,
                  const float* __restrict__ b1,
                  const float* __restrict__ b2,
                  const float* __restrict__ bank,
                  const char* __restrict__ ws,
                  const float* __restrict__ mn,
                  float* __restrict__ out) {
    extern __shared__ __align__(32) char smem_raw[];
    const int tid  = threadIdx.x;
    const int lane = tid & 31;
    const int wave = tid >> 5;
    const int m    = lane & 15;       // column (B/C/D) or row (A) index within tile
    const int half = lane >> 4;

    __bf16* stageH  = (__bf16*)(smem_raw + wave * WAVE_SMEM);
    __bf16* stageQ  = (__bf16*)(smem_raw + wave * WAVE_SMEM + 16384);
    __bf16* stageFF = (__bf16*)(smem_raw + wave * WAVE_SMEM + 24576);
    char*   chunk   = smem_raw + CHUNK_OFF;          // two SUB_SZ buffers

    const int row0 = blockIdx.x * 128 + wave * 16;

    // stage bmatT (16KB) + w1T (32KB) across both pipeline buffers (adjacent in ws)
    issue_async(chunk, ws + WS_BMATT, tid, 16*512*2 + 512*32*2);
    wait_async_zero();
    __syncthreads();
    const __bf16* ldsBmat = (const __bf16*)chunk;
    const __bf16* ldsW1   = (const __bf16*)(chunk + 16*512*2);

    // ---- GEMM1: xp[16x16] = query[16x512] @ B_mat (cols >=10 zero-padded) ----
    f32x8 c1 = {};
    {
        const float* qrow = query + (size_t)(row0 + m) * D_DIM;
        for (int kt = 0; kt < D_DIM / 32; ++kt) {
            bf16x16 a;
            const int kb = kt * 32 + half * 8;
            #pragma unroll
            for (int i = 0; i < 8; ++i) {
                a[i]     = (__bf16)qrow[kb + i];
                a[8 + i] = (__bf16)qrow[kb + 16 + i];
            }
            bf16x16 bfr = load_b_frag(ldsBmat, D_DIM, m, kt * 32, half);
            c1 = __builtin_amdgcn_wmma_f32_16x16x32_bf16(false, a, false, bfr, (short)0, c1, false, false);
        }
    }

    // ---- Fourier features: ff = [sin(xp), cos(xp)], staged 16x32 bf16 (K 20..31 zero) ----
    {   // one 32B store per lane zeroes the whole 16x32 tile
        bf16x16 z = {};
        *(bf16x16*)(stageFF + lane * 16) = z;
    }
    __syncthreads();
    if (m < F_DIM) {
        #pragma unroll
        for (int r = 0; r < 8; ++r) {
            const int mm = r + half * 8;          // D layout: lanes 0-15 rows 0-7, lanes 16-31 rows 8-15
            const float x = c1[r];
            stageFF[mm * 32 + m]         = (__bf16)__sinf(x);
            stageFF[mm * 32 + F_DIM + m] = (__bf16)__cosf(x);
        }
    }
    __syncthreads();

    // ---- GEMM2: h[16x512] = relu(ff @ W1 + b1), staged bf16 ----
    {
        const bf16x16 aff = load_a_lds(stageFF, 32, m, 0, half);
        for (int nt = 0; nt < H_DIM / 16; ++nt) {
            bf16x16 bfr = load_b_frag(ldsW1, 32, nt * 16 + m, 0, half);
            f32x8 c = {};
            c = __builtin_amdgcn_wmma_f32_16x16x32_bf16(false, aff, false, bfr, (short)0, c, false, false);
            const int col = nt * 16 + m;
            const float bias = b1[col];
            #pragma unroll
            for (int r = 0; r < 8; ++r) {
                const int mm = r + half * 8;
                float v = fmaxf(c[r] + bias, 0.0f);
                stageH[mm * H_DIM + col] = (__bf16)v;
            }
        }
    }
    __syncthreads();   // all waves done with bmat/W1 panel before it is overwritten

    // ---- GEMM3: q[16x256] = h @ W2 + b2; W2^T pipelined in 8 x 32KB sub-chunks ----
    float qn2[8];
    #pragma unroll
    for (int r = 0; r < 8; ++r) qn2[r] = 0.0f;

    issue_async(chunk, ws + WS_W2T, tid, SUB_SZ);     // prefetch sub-chunk 0
    wait_async_zero();
    __syncthreads();
    for (int s = 0; s < 8; ++s) {                     // 32 cols x 512 K per sub-chunk
        const __bf16* ldsW2 = (const __bf16*)(chunk + (s & 1) * SUB_SZ);
        if (s + 1 < 8)                                 // overlap DMA of s+1 with compute of s
            issue_async(chunk + ((s + 1) & 1) * SUB_SZ,
                        ws + WS_W2T + (s + 1) * SUB_SZ, tid, SUB_SZ);
        for (int nt2 = 0; nt2 < 2; ++nt2) {
            const int nt = s * 2 + nt2;
            f32x8 c = {};
            for (int kt = 0; kt < H_DIM / 32; ++kt) {
                bf16x16 a   = load_a_lds(stageH, H_DIM, m, kt * 32, half);
                bf16x16 bfr = load_b_frag(ldsW2, H_DIM, nt2 * 16 + m, kt * 32, half);
                c = __builtin_amdgcn_wmma_f32_16x16x32_bf16(false, a, false, bfr, (short)0, c, false, false);
            }
            const int col = nt * 16 + m;
            const float bias = b2[col];
            #pragma unroll
            for (int r = 0; r < 8; ++r) {
                const int mm = r + half * 8;
                const float v = c[r] + bias;
                qn2[r] += v * v;
                stageQ[mm * M_DIM + col] = (__bf16)v;
            }
        }
        wait_async_zero();
        __syncthreads();                               // next buffer staged; readers of cur done
    }
    // reduce ||q||^2 across the 16 lanes holding each row's columns
    #pragma unroll
    for (int xm = 1; xm < 16; xm <<= 1) {
        #pragma unroll
        for (int r = 0; r < 8; ++r) qn2[r] += __shfl_xor(qn2[r], xm, 32);
    }
    float qn[8];
    #pragma unroll
    for (int r = 0; r < 8; ++r) qn[r] = sqrtf(qn2[r]);

    // ---- GEMM4: sim = q @ bank^T / max(|q||m|,eps); bank^T pipelined in 16 x 32KB ----
    float maxv[8];
    int   maxi[8];
    #pragma unroll
    for (int r = 0; r < 8; ++r) { maxv[r] = -3.402823466e38f; maxi[r] = 0; }

    issue_async(chunk, ws + WS_BANKT, tid, SUB_SZ);   // prefetch sub-chunk 0
    wait_async_zero();
    __syncthreads();
    for (int s = 0; s < 16; ++s) {                    // 64 bank rows x 256 K per sub-chunk
        const __bf16* ldsBank = (const __bf16*)(chunk + (s & 1) * SUB_SZ);
        if (s + 1 < 16)
            issue_async(chunk + ((s + 1) & 1) * SUB_SZ,
                        ws + WS_BANKT + (s + 1) * SUB_SZ, tid, SUB_SZ);
        for (int nt4 = 0; nt4 < 4; ++nt4) {
            const int nt = s * 4 + nt4;
            f32x8 c = {};
            for (int kt = 0; kt < M_DIM / 32; ++kt) {
                bf16x16 a   = load_a_lds(stageQ, M_DIM, m, kt * 32, half);
                bf16x16 bfr = load_b_frag(ldsBank, M_DIM, nt4 * 16 + m, kt * 32, half);
                c = __builtin_amdgcn_wmma_f32_16x16x32_bf16(false, a, false, bfr, (short)0, c, false, false);
            }
            const int col = nt * 16 + m;
            const float mnc = mn[col];
            #pragma unroll
            for (int r = 0; r < 8; ++r) {
                const float sim = c[r] / fmaxf(qn[r] * mnc, SIM_EPS);
                if (sim > maxv[r] || (sim == maxv[r] && col < maxi[r])) { maxv[r] = sim; maxi[r] = col; }
            }
        }
        wait_async_zero();
        __syncthreads();
    }
    // butterfly argmax across 16 lanes (halves stay separate: masks < 16)
    #pragma unroll
    for (int xm = 1; xm < 16; xm <<= 1) {
        #pragma unroll
        for (int r = 0; r < 8; ++r) {
            const float ov = __shfl_xor(maxv[r], xm, 32);
            const int   oi = __shfl_xor(maxi[r], xm, 32);
            if (ov > maxv[r] || (ov == maxv[r] && oi < maxi[r])) { maxv[r] = ov; maxi[r] = oi; }
        }
    }

    // ---- threshold + gather: out[row] = (max_sim >= 0.7) ? bank[argmax] : 0 ----
    #pragma unroll
    for (int r = 0; r < 8; ++r) {
        const int mm = r + half * 8;
        const size_t row = (size_t)(row0 + mm);
        const bool pass = maxv[r] >= THRESH;
        const float* src = bank + (size_t)maxi[r] * M_DIM;
        float* dst = out + row * M_DIM;
        #pragma unroll
        for (int jt = 0; jt < 4; ++jt) {              // 16 lanes x float4 = 64 floats/step
            const int j = jt * 64 + m * 4;
            float4 v;
            if (pass) v = *(const float4*)(src + j);
            else      v = make_float4(0.f, 0.f, 0.f, 0.f);
            *(float4*)(dst + j) = v;
        }
    }
}

extern "C" void kernel_launch(void* const* d_in, const int* in_sizes, int n_in,
                              void* d_out, int out_size, void* d_ws, size_t ws_size,
                              hipStream_t stream) {
    const float* query = (const float*)d_in[0];
    const float* Bm    = (const float*)d_in[1];
    const float* W1    = (const float*)d_in[2];
    const float* b1    = (const float*)d_in[3];
    const float* W2    = (const float*)d_in[4];
    const float* b2    = (const float*)d_in[5];
    const float* bank  = (const float*)d_in[6];

    char* ws = (char*)d_ws;
    __bf16* bmatT = (__bf16*)(ws + WS_BMATT);
    __bf16* w1T   = (__bf16*)(ws + WS_W1T);
    __bf16* w2T   = (__bf16*)(ws + WS_W2T);
    __bf16* bankT = (__bf16*)(ws + WS_BANKT);
    float*  mn    = (float*)(ws + WS_MN);

    prep_weights<<<512, 256, 0, stream>>>(Bm, W1, W2, bank, bmatT, w1T, w2T, bankT);
    prep_mn<<<N_MEM / 8, 256, 0, stream>>>(bank, mn);

    immune_fused<<<BATCH / 128, 256, TOTAL_SMEM, stream>>>(
        query, b1, b2, bank, ws, mn, (float*)d_out);
}
